// AttentionSimilarity_39135742001557
// MI455X (gfx1250) — matrix operand order repaired and checked by reading, not compiled, not run on hardware
//
#include <hip/hip_runtime.h>
#include <hip/hip_bf16.h>
#include <math.h>

typedef __attribute__((ext_vector_type(16))) __bf16 v16bf;
typedef __attribute__((ext_vector_type(8)))  float  v8f;

constexpr int BATCH = 128;
constexpr int CH    = 768;
constexpr int SEQ   = 49;
constexpr int SP    = 64;     // padded seq
constexpr int INNER = 96;
constexpr int ROWS  = BATCH * SEQ;           // 6272
constexpr size_t NPROJ = (size_t)ROWS * INNER;
constexpr int VTSTRIDE = INNER * SP;         // 6144 elems per batch
constexpr float  EPSC  = 1e-8f;

// ---- workspace layout (bytes, all 16B aligned) ----
constexpr size_t SZ_W1T  = (size_t)3 * CH * CH * 2;          // 3,538,944
constexpr size_t SZ_W2T  = (size_t)3 * INNER * CH * 2;       //   442,368
constexpr size_t SZ_QKB  = (size_t)6 * NPROJ * 2;            // 7,225,344
constexpr size_t SZ_VT   = (size_t)2 * BATCH * VTSTRIDE * 2; // 3,145,728
constexpr size_t SZ_XB   = (size_t)2 * ROWS * CH * 2;        // 19,267,584
constexpr size_t OFF_W1T = 0;
constexpr size_t OFF_W2T = OFF_W1T + SZ_W1T;
constexpr size_t OFF_QKB = OFF_W2T + SZ_W2T;
constexpr size_t OFF_VT  = OFF_QKB + SZ_QKB;
constexpr size_t OFF_XB  = OFF_VT + SZ_VT;

// ---------------- CDNA5 async global->LDS helpers ----------------
__device__ inline uint32_t lds_off(const void* p) {
  // generic shared pointer: low 32 bits are the LDS byte address (ISA 10.2)
  return (uint32_t)(uintptr_t)p;
}
__device__ inline void async_b128(uint32_t lds, const void* gsrc) {
  asm volatile("global_load_async_to_lds_b128 %0, %1, off"
               :: "v"(lds), "v"((uint64_t)(uintptr_t)gsrc) : "memory");
}
__device__ inline void wait_async0() {
#if __has_builtin(__builtin_amdgcn_s_wait_asynccnt)
  __builtin_amdgcn_s_wait_asynccnt(0);
#else
  asm volatile("s_wait_asynccnt 0" ::: "memory");
#endif
}
__device__ inline void wait_async_le2() {   // allow the 2 next-buffer DMAs in flight
#if __has_builtin(__builtin_amdgcn_s_wait_asynccnt)
  __builtin_amdgcn_s_wait_asynccnt(2);
#else
  asm volatile("s_wait_asynccnt 2" ::: "memory");
#endif
}
// flat contiguous async copy (bytes % 16 == 0), all 256 threads participate
__device__ inline void async_copy(uint32_t lds, const void* gsrc, int bytes) {
  const char* g = (const char*)gsrc;
  for (int e = threadIdx.x * 16; e < bytes; e += 256 * 16)
    async_b128(lds + e, g + e);
}

// ---------------- WMMA helpers (CDNA5 16x16x32 bf16, f32 acc) ----------------
__device__ inline v8f wmma_bf16(v16bf a, v16bf b, v8f c) {
  return __builtin_amdgcn_wmma_f32_16x16x32_bf16(false, a, false, b, (short)0, c,
                                                 false, false);
}
// A fragment: 16x32; row m, VGPR j holds K pair kb = (j<4 ? 2j : 2j+8) + 8h
__device__ inline v16bf load_a_frag(const __bf16* base, int row, int stride,
                                    int k0, int h) {
  v16bf a;
#pragma unroll
  for (int j = 0; j < 8; ++j) {
    int kb = ((j < 4) ? (2 * j) : (2 * j + 8)) + 8 * h;
    a[2 * j]     = base[row * stride + k0 + kb];
    a[2 * j + 1] = base[row * stride + k0 + kb + 1];
  }
  return a;
}
// B fragment: 32x16; col n, VGPR j holds K pair kb = (j<4 ? 2j : 2j+8) + 8h
__device__ inline v16bf load_b_frag(const __bf16* base, int col, int stride,
                                    int k0, int h) {
  v16bf b;
#pragma unroll
  for (int j = 0; j < 8; ++j) {
    int kb = ((j < 4) ? (2 * j) : (2 * j + 8)) + 8 * h;
    b[2 * j]     = base[col * stride + k0 + kb];
    b[2 * j + 1] = base[col * stride + k0 + kb + 1];
  }
  return b;
}

// ---------------- kernel 0: weight prep + vt padding clear ----------------
__global__ __launch_bounds__(256) void prep_weights(
    const float* Wq1, const float* Wq2, const float* Wk1, const float* Wk2,
    const float* Wv1, const float* Wv2, __bf16* w1t, __bf16* w2t, __bf16* vtb) {
  const float* W1[3] = {Wq1, Wk1, Wv1};
  const float* W2[3] = {Wq2, Wk2, Wv2};
  const int N1 = 3 * CH * CH;
  const int N2 = 3 * INNER * CH;
  const int N3 = 2 * BATCH * VTSTRIDE;
  int idx = blockIdx.x * 256 + threadIdx.x;
  if (idx < N1) {
    int t = idx / (CH * CH);
    int rem = idx - t * CH * CH;
    int n = rem / CH, k = rem - n * CH;
    w1t[idx] = (__bf16)W1[t][k * CH + n];            // w1t[t][n][k] = W1[k][n]
  } else if (idx < N1 + N2) {
    int j = idx - N1;
    int t = j / (INNER * CH);
    int rem = j - t * INNER * CH;
    int n = rem / CH, k = rem - n * CH;
    w2t[j] = (__bf16)W2[t][k * INNER + n];           // w2t[t][n][k] = W2[k][n]
  } else if (idx < N1 + N2 + N3) {
    vtb[idx - N1 - N2] = (__bf16)0.0f;               // zero v^T (padding stays 0)
  }
}

// ---------------- kernel 0b: feature transpose fp32[B,C,S] -> bf16[row][C] ----------------
__global__ __launch_bounds__(256) void xpose_kernel(const float* fa,
                                                    const float* fb, __bf16* xb) {
  const size_t NF = (size_t)BATCH * CH * SEQ;        // per feature set
  size_t idx = (size_t)blockIdx.x * 256 + threadIdx.x;
  if (idx >= 2 * NF) return;
  int f = (idx >= NF) ? 1 : 0;
  size_t r = idx - (size_t)f * NF;                   // = (b*CH + c)*SEQ + s
  int s = (int)(r % SEQ);
  size_t bc = r / SEQ;
  int c = (int)(bc % CH);
  int b = (int)(bc / CH);
  float v = (f ? fb : fa)[r];                        // contiguous read
  xb[(size_t)f * ROWS * CH + ((size_t)b * SEQ + s) * CH + c] = (__bf16)v;
}

// ---------------- projection staging (wave-uniform: 2 async instr / wave) ----------------
__device__ inline void stage_g1(const __bf16* w1nc, const __bf16* xrow,
                                uint32_t buf, int k0, int w, int ln) {
  {  // Ws [64n][32k]: 256 x 16B chunks, 32 per wave
    int e = w * 32 + ln;
    int n = e >> 2, sub = e & 3;
    async_b128(buf + e * 16, w1nc + (size_t)n * CH + k0 + sub * 8);
  }
  {  // Xs [32m][32k]: 128 x 16B chunks, 16 per wave (half-masked lanes)
    int e = w * 16 + (ln & 15);
    int m = e >> 2, sub = e & 3;
    if (ln < 16)
      async_b128(buf + 4096 + e * 16, xrow + (size_t)m * CH + k0 + sub * 8);
  }
}
__device__ inline void stage_g2(const __bf16* w2t, uint32_t buf, int k0,
                                int w, int ln) {
  // W2s [96n][32k]: 384 x 16B chunks, 48 per wave
  {
    int e = w * 48 + ln;
    int n = e >> 2, sub = e & 3;
    async_b128(buf + e * 16, w2t + (size_t)n * CH + k0 + sub * 8);
  }
  {
    int e = w * 48 + 32 + (ln & 15);
    int n = e >> 2, sub = e & 3;
    if (ln < 16)
      async_b128(buf + e * 16, w2t + (size_t)n * CH + k0 + sub * 8);
  }
}

// ---------------- kernel 1: fused projection ReLU(X@W1)@W2 -> bf16 ----------------
// grid = (6272/32, 6); block = 256 (8 waves)
__global__ __launch_bounds__(256) void proj_kernel(
    const __bf16* xb, const __bf16* w1t_all, const __bf16* w2t_all,
    __bf16* qkb, __bf16* vtb) {
  __shared__ __bf16 Hs[32 * 768];                       // 48 KB
  __shared__ __align__(16) unsigned char sarena[2][6144]; // 12 KB staging (double buf)

  const int tid = threadIdx.x;
  const int w = tid >> 5, ln = tid & 31;
  const int l15 = ln & 15, h = ln >> 4;

  const int p = blockIdx.y;             // f*3 + t
  const int f = p / 3, t = p - 3 * f;
  const __bf16* w1t = w1t_all + (size_t)t * CH * CH;
  const __bf16* w2t = w2t_all + (size_t)t * INNER * CH;
  const int rowBase = blockIdx.x * 32;
  const __bf16* xrow = xb + (size_t)f * ROWS * CH + (size_t)rowBase * CH;
  const uint32_t sb0 = lds_off(sarena[0]);
  const uint32_t sb1 = lds_off(sarena[1]);

  // ---- GEMM1 + ReLU -> Hs ----
  const int mi1 = w >> 2, ni1 = w & 3;  // 2x4 tiles per 64-wide N chunk
  for (int nc = 0; nc < 12; ++nc) {
    const __bf16* w1nc = w1t + (size_t)(nc * 64) * CH;
    v8f acc = {};
    stage_g1(w1nc, xrow, sb0, 0, w, ln);
    for (int kit = 0; kit < 24; ++kit) {
      const int cur = kit & 1;
      if (kit + 1 < 24) {                       // prefetch next k-slice
        stage_g1(w1nc, xrow, cur ? sb0 : sb1, (kit + 1) * 32, w, ln);
        wait_async_le2();
      } else {
        wait_async0();
      }
      __syncthreads();
      const __bf16* Wsb = (const __bf16*)sarena[cur];
      const __bf16* Xsb = (const __bf16*)(sarena[cur] + 4096);
      v16bf afrag = load_a_frag(Xsb, 16 * mi1 + l15, 32, 0, h);
      v16bf bfrag = load_b_frag(Wsb, 16 * ni1 + l15, 32, 0, h);
      acc = wmma_bf16(afrag, bfrag, acc);
      __syncthreads();
    }
#pragma unroll
    for (int j = 0; j < 8; ++j) {       // D: m = 16*mi + j + 8h, n = 16*ni + l15
      int m = 16 * mi1 + j + 8 * h;
      int n = nc * 64 + 16 * ni1 + l15;
      Hs[m * 768 + n] = (__bf16)fmaxf(acc[j], 0.0f);
    }
  }
  __syncthreads();

  // ---- GEMM2: Hs @ W2^T ----
  const int mi2 = w & 1;
  const int nia = w >> 1;
  const bool two = (w < 4);
  const int nib = (w >> 1) + 4;
  v8f acc0 = {}, acc1 = {};
  stage_g2(w2t, sb0, 0, w, ln);
  for (int kit = 0; kit < 24; ++kit) {
    const int cur = kit & 1;
    if (kit + 1 < 24) {
      stage_g2(w2t, cur ? sb0 : sb1, (kit + 1) * 32, w, ln);
      wait_async_le2();
    } else {
      wait_async0();
    }
    __syncthreads();
    const __bf16* W2sb = (const __bf16*)sarena[cur];
    v16bf afrag = load_a_frag(Hs, 16 * mi2 + l15, 768, kit * 32, h);
    v16bf b0 = load_b_frag(W2sb, 16 * nia + l15, 32, 0, h);
    acc0 = wmma_bf16(afrag, b0, acc0);
    if (two) {                          // wave-uniform
      v16bf b1 = load_b_frag(W2sb, 16 * nib + l15, 32, 0, h);
      acc1 = wmma_bf16(afrag, b1, acc1);
    }
    __syncthreads();
  }

  // epilogue: q/k -> row-major bf16 [6272][96]; v -> transposed padded [batch][96][64]
  if (t < 2) {
    __bf16* outb = qkb + (size_t)p * NPROJ;
#pragma unroll
    for (int j = 0; j < 8; ++j) {
      int m = 16 * mi2 + j + 8 * h;
      outb[(size_t)(rowBase + m) * INNER + 16 * nia + l15] = (__bf16)acc0[j];
    }
    if (two) {
#pragma unroll
      for (int j = 0; j < 8; ++j) {
        int m = 16 * mi2 + j + 8 * h;
        outb[(size_t)(rowBase + m) * INNER + 16 * nib + l15] = (__bf16)acc1[j];
      }
    }
  } else {
    __bf16* vt = vtb + (size_t)f * BATCH * VTSTRIDE;
#pragma unroll
    for (int j = 0; j < 8; ++j) {
      int row = rowBase + 16 * mi2 + j + 8 * h;
      int b = row / SEQ, s = row - b * SEQ;
      vt[(size_t)b * VTSTRIDE + (16 * nia + l15) * SP + s] = (__bf16)acc0[j];
    }
    if (two) {
#pragma unroll
      for (int j = 0; j < 8; ++j) {
        int row = rowBase + 16 * mi2 + j + 8 * h;
        int b = row / SEQ, s = row - b * SEQ;
        vt[(size_t)b * VTSTRIDE + (16 * nib + l15) * SP + s] = (__bf16)acc1[j];
      }
    }
  }
}

// ---------------- kernel 2: pairwise attention + cosine ----------------
// grid = (128 aa, 128 bb); block = 256 (8 waves)
__global__ __launch_bounds__(256) void attn_kernel(const __bf16* qkb,
                                                   const __bf16* vtb,
                                                   float* out) {
  constexpr int OQ  = 0;                       // bf16 [64][96]
  constexpr int OKK = 12288;                   // bf16 [64][96]
  constexpr int OVA = 24576;                   // bf16 [96][64]
  constexpr int OVB = 36864;                   // bf16 [96][64]
  constexpr int OS  = 49152;                   // f32 [49][64]; probs bf16[64][64] overlays
  constexpr int ODT = OS + 49 * 64 * 4;
  constexpr int ONA = ODT + 256;
  constexpr int ONV = ONA + 256;
  constexpr int OCT = ONV + 256;
  __shared__ __align__(16) unsigned char arena[OCT + 16];   // 62.5 KB

  __bf16* Q    = (__bf16*)(arena + OQ);
  __bf16* Kt   = (__bf16*)(arena + OKK);
  __bf16* VA   = (__bf16*)(arena + OVA);
  __bf16* VB   = (__bf16*)(arena + OVB);
  float*  Sm   = (float*)(arena + OS);
  __bf16* P    = (__bf16*)(arena + OS);
  float*  dotS = (float*)(arena + ODT);
  float*  naS  = (float*)(arena + ONA);
  float*  nvS  = (float*)(arena + ONV);
  float*  cosT = (float*)(arena + OCT);

  const int tid = threadIdx.x;
  const int w = tid >> 5, ln = tid & 31;
  const int l15 = ln & 15, h = ln >> 4;
  const int aa = blockIdx.x, bb = blockIdx.y;

  const __bf16* q_a = qkb + 0 * NPROJ;
  const __bf16* k_a = qkb + 1 * NPROJ;
  const __bf16* q_b = qkb + 3 * NPROJ;
  const __bf16* k_b = qkb + 4 * NPROJ;
  const __bf16* vt_a = vtb;
  const __bf16* vt_b = vtb + (size_t)BATCH * VTSTRIDE;

  constexpr int QKBYTES = SEQ * INNER * 2;     // 9408 (588 x 16B)
  constexpr int VTBYTES = INNER * SP * 2;      // 12288

  if (tid == 0) *cosT = 0.0f;
  // zero padded rows 49..63 of Q and K once (restage only rewrites rows < 49)
  for (int e = tid; e < (SP - SEQ) * INNER; e += 256) {
    Q[SEQ * INNER + e]  = (__bf16)0.0f;
    Kt[SEQ * INNER + e] = (__bf16)0.0f;
  }
  // phase 0 (BA): scores = qb @ ka^T ; aligned_a = probs @ va ; cmp vb
  async_copy(lds_off(Q),  q_b + (size_t)bb * SEQ * INNER, QKBYTES);
  async_copy(lds_off(Kt), k_a + (size_t)aa * SEQ * INNER, QKBYTES);
  async_copy(lds_off(VA), vt_a + (size_t)aa * VTSTRIDE, VTBYTES);
  async_copy(lds_off(VB), vt_b + (size_t)bb * VTSTRIDE, VTBYTES);

  const float scale = 0.1020620726159657f;     // 1/sqrt(96)
  float ex[SEQ];

  for (int phase = 0; phase < 2; ++phase) {
    const __bf16* Vop  = (phase == 0) ? VA : VB;  // B operand of aligned GEMM
    const __bf16* Vcmp = (phase == 0) ? VB : VA;  // cosine comparison

    if (tid < SP) { dotS[tid] = 0.0f; naS[tid] = 0.0f; nvS[tid] = 0.0f; }
    wait_async0();
    __syncthreads();

    if (tid < SEQ) {                            // ||v_cmp[row]||^2
      float s = 0.0f;
      for (int n = 0; n < INNER; ++n) {
        float v = (float)Vcmp[n * SP + tid];
        s += v * v;
      }
      nvS[tid] = s;
    }

    // ---- scores: [64x64] = Q @ K^T over K=96
    {
      const int ni = w & 3;
      const int miA = w >> 2, miB = miA + 2;
      v8f s0 = {}, s1 = {};
      for (int kit = 0; kit < 3; ++kit) {
        int k0 = kit * 32;
        v16bf bfrag = load_b_frag(Kt, 16 * ni + l15, INNER, k0, h);
        v16bf a0 = load_a_frag(Q, 16 * miA + l15, INNER, k0, h);
        s0 = wmma_bf16(a0, bfrag, s0);
        v16bf a1 = load_a_frag(Q, 16 * miB + l15, INNER, k0, h);
        s1 = wmma_bf16(a1, bfrag, s1);
      }
#pragma unroll
      for (int j = 0; j < 8; ++j) {
        int n = 16 * ni + l15;
        int mA = 16 * miA + j + 8 * h;
        if (mA < SEQ) Sm[mA * SP + n] = s0[j] * scale;
        int mB = 16 * miB + j + 8 * h;
        if (mB < SEQ) Sm[mB * SP + n] = s1[j] * scale;
      }
    }
    __syncthreads();                            // Q/Kt now dead for this phase

    if (phase == 0) {                           // overlap phase-1 Q/K DMA with
      async_copy(lds_off(Q),  q_a + (size_t)aa * SEQ * INNER, QKBYTES);  // softmax+
      async_copy(lds_off(Kt), k_b + (size_t)bb * SEQ * INNER, QKBYTES);  // aligned GEMM
    }

    // ---- softmax over 49 keys into registers (probs overlays Sm)
    float inv = 0.0f;
    if (tid < SEQ) {
      float mx = -1e30f;
#pragma unroll
      for (int j = 0; j < SEQ; ++j) mx = fmaxf(mx, Sm[tid * SP + j]);
      float sum = 0.0f;
#pragma unroll
      for (int j = 0; j < SEQ; ++j) {
        float e = __expf(Sm[tid * SP + j] - mx);
        ex[j] = e;
        sum += e;
      }
      inv = 1.0f / sum;
    }
    __syncthreads();
    if (tid < SEQ) {
#pragma unroll
      for (int j = 0; j < SEQ; ++j) P[tid * SP + j] = (__bf16)(ex[j] * inv);
      for (int j = SEQ; j < SP; ++j) P[tid * SP + j] = (__bf16)0.0f;
    } else if (tid < SP) {
      for (int j = 0; j < SP; ++j) P[tid * SP + j] = (__bf16)0.0f;
    }
    __syncthreads();

    // ---- aligned = P @ v : [64x96] over K=64
    {
      const int miP = w & 3;
      const int nb = w >> 2;
      v8f al0 = {}, al1 = {}, al2 = {};
      for (int kit = 0; kit < 2; ++kit) {
        int k0 = kit * 32;
        v16bf afrag = load_a_frag(P, 16 * miP + l15, SP, k0, h);
        v16bf b0 = load_b_frag(Vop, 16 * nb + l15, SP, k0, h);
        al0 = wmma_bf16(afrag, b0, al0);
        v16bf b1 = load_b_frag(Vop, 16 * (nb + 2) + l15, SP, k0, h);
        al1 = wmma_bf16(afrag, b1, al1);
        v16bf b2 = load_b_frag(Vop, 16 * (nb + 4) + l15, SP, k0, h);
        al2 = wmma_bf16(afrag, b2, al2);
      }
#pragma unroll
      for (int j = 0; j < 8; ++j) {             // cosine partials via ds_add_f32
        int m = 16 * miP + j + 8 * h;
        if (m < SEQ) {
          int n0 = 16 * nb + l15;
          int n1 = 16 * (nb + 2) + l15;
          int n2 = 16 * (nb + 4) + l15;
          float a0 = al0[j], a1 = al1[j], a2 = al2[j];
          atomicAdd(&dotS[m], a0 * (float)Vcmp[n0 * SP + m] +
                              a1 * (float)Vcmp[n1 * SP + m] +
                              a2 * (float)Vcmp[n2 * SP + m]);
          atomicAdd(&naS[m], a0 * a0 + a1 * a1 + a2 * a2);
        }
      }
    }
    __syncthreads();

    if (tid < SEQ) {
      float na = fmaxf(sqrtf(naS[tid]), EPSC);
      float nv = fmaxf(sqrtf(nvS[tid]), EPSC);
      atomicAdd(cosT, dotS[tid] / (na * nv));
    }
    __syncthreads();
  }

  if (tid == 0) out[(size_t)bb * BATCH + aa] = *cosT / (float)SEQ;
}

// ---------------- launcher ----------------
extern "C" void kernel_launch(void* const* d_in, const int* in_sizes, int n_in,
                              void* d_out, int out_size, void* d_ws, size_t ws_size,
                              hipStream_t stream) {
  (void)in_sizes; (void)n_in; (void)out_size; (void)ws_size;
  const float* fa  = (const float*)d_in[0];
  const float* fb  = (const float*)d_in[1];
  const float* Wq1 = (const float*)d_in[2];
  const float* Wq2 = (const float*)d_in[3];
  const float* Wk1 = (const float*)d_in[4];
  const float* Wk2 = (const float*)d_in[5];
  const float* Wv1 = (const float*)d_in[6];
  const float* Wv2 = (const float*)d_in[7];

  unsigned char* ws = (unsigned char*)d_ws;
  __bf16* w1t = (__bf16*)(ws + OFF_W1T);
  __bf16* w2t = (__bf16*)(ws + OFF_W2T);
  __bf16* qkb = (__bf16*)(ws + OFF_QKB);
  __bf16* vtb = (__bf16*)(ws + OFF_VT);
  __bf16* xb  = (__bf16*)(ws + OFF_XB);
  float*  out = (float*)d_out;

  const int NPREP = 3 * CH * CH + 3 * INNER * CH + 2 * BATCH * VTSTRIDE;
  prep_weights<<<(NPREP + 255) / 256, 256, 0, stream>>>(Wq1, Wq2, Wk1, Wk2,
                                                        Wv1, Wv2, w1t, w2t, vtb);
  const size_t NX = (size_t)2 * BATCH * CH * SEQ;
  xpose_kernel<<<(unsigned)((NX + 255) / 256), 256, 0, stream>>>(fa, fb, xb);
  proj_kernel<<<dim3(ROWS / 32, 6), 256, 0, stream>>>(xb, w1t, w2t, qkb, vtb);
  attn_kernel<<<dim3(BATCH, BATCH), 256, 0, stream>>>(qkb, vtb, out);
}
